// ROEN_Fast_Transformer_24927990186454
// MI455X (gfx1250) — matrix operand order repaired and checked by reading, not compiled, not run on hardware
//
#include <hip/hip_runtime.h>
#include <math.h>

// ---------------- problem constants ----------------
constexpr int kT = 8;
constexpr int kN = 10000;
constexpr int kE = 60000;
constexpr int kNodeIn = 64;
constexpr int kEdgeIn = 32;
constexpr int kH = 128;
constexpr int kNH = 4;
constexpr int kNC = 5;
constexpr int kRows = kT * kN; // 80000
constexpr float kInvSqrtHD = 0.17677669529663687f; // 1/sqrt(32)

typedef __attribute__((ext_vector_type(16))) _Float16 v16h;
typedef __attribute__((ext_vector_type(8)))  float    v8f;

// ---------------- WMMA fragment helpers ----------------
__device__ __forceinline__ v16h frag_a_f16(const _Float16* row, int k0, int lane) {
  int hsel = (lane >> 4) & 1;
  v16h a;
#pragma unroll
  for (int i = 0; i < 8; ++i) {
    int kk = k0 + hsel * 16 + 2 * i;
    a[2 * i]     = row[kk];
    a[2 * i + 1] = row[kk + 1];
  }
  return a;
}
__device__ __forceinline__ v16h frag_a_f32(const float* row, int k0, int lane) {
  int hsel = (lane >> 4) & 1;
  v16h a;
#pragma unroll
  for (int i = 0; i < 8; ++i) {
    int kk = k0 + hsel * 16 + 2 * i;
    a[2 * i]     = (_Float16)row[kk];
    a[2 * i + 1] = (_Float16)row[kk + 1];
  }
  return a;
}
// B packed layout: Bp[((kb*NT + nt)*32 + lane)*16 + j] = B[kb*32 + (lane>=16?16:0) + j][nt*16 + (lane&15)]
__device__ __forceinline__ v16h frag_b_packed(const _Float16* Bp, int ntTiles, int kb, int nt, int lane) {
  const _Float16* p = Bp + ((size_t)(kb * ntTiles + nt) * 32 + lane) * 16;
  v16h b;
#pragma unroll
  for (int j = 0; j < 16; ++j) b[j] = p[j];
  return b;
}
__device__ __forceinline__ v8f wmma32(v16h a, v16h b, v8f c) {
  return __builtin_amdgcn_wmma_f32_16x16x32_f16(false, a, false, b, (short)0, c, false, false);
}

// ---------------- weight packing ----------------
__global__ void k_pack_b(const float* W, _Float16* out, int Krows, int Ncols, int KB, int NT) {
  long long tid = (long long)blockIdx.x * blockDim.x + threadIdx.x;
  long long total = (long long)KB * NT * 512;
  if (tid >= total) return;
  int j = tid & 15;
  int lane = (tid >> 4) & 31;
  int rest = (int)(tid >> 9);
  int nt = rest % NT, kb = rest / NT;
  int kk = kb * 32 + ((lane >> 4) & 1) * 16 + j;
  int col = nt * 16 + (lane & 15);
  float v = (kk < Krows && col < Ncols) ? W[(size_t)kk * Ncols + col] : 0.f;
  out[tid] = (_Float16)v;
}
// conv tap: B[ci][co] = W[co,ci,0,j]  (W is [32,128,1,ksz]); Krows=128, Ncols=32 -> KB=4, NT=2
__global__ void k_pack_conv(const float* W, _Float16* out, int ksz, int j) {
  int tid = blockIdx.x * blockDim.x + threadIdx.x;
  if (tid >= 4 * 2 * 512) return;
  int jj = tid & 15;
  int lane = (tid >> 4) & 31;
  int rest = tid >> 9;
  int nt = rest % 2, kb = rest / 2;
  int ci = kb * 32 + ((lane >> 4) & 1) * 16 + jj;
  int co = nt * 16 + (lane & 15);
  out[tid] = (_Float16)W[((size_t)co * 128 + ci) * ksz + j];
}

__global__ void k_fill(unsigned* p, unsigned v, long long n) {
  long long i = (long long)blockIdx.x * blockDim.x + threadIdx.x;
  if (i < n) p[i] = v;
}

// ---------------- strip GEMM kernels: one wave owns a 16x128 output strip ----------------
// h = relu(X[M,64] @ Wenc + b) -> f32
__global__ void k_node_enc(const float* X, const _Float16* Bp, const float* bias, float* out, int M) {
  int wid = blockIdx.x * (blockDim.x >> 5) + (threadIdx.x >> 5);
  int lane = threadIdx.x & 31;
  if (wid * 16 >= M) return;
  const float* row = X + (size_t)(wid * 16 + (lane & 15)) * 64;
  v8f acc[8] = {};
#pragma unroll
  for (int kb = 0; kb < 2; ++kb) {
    v16h a = frag_a_f32(row, kb * 32, lane);
#pragma unroll
    for (int nt = 0; nt < 8; ++nt)
      acc[nt] = wmma32(a, frag_b_packed(Bp, 8, kb, nt, lane), acc[nt]);
  }
  int mb = wid * 16 + ((lane < 16) ? 0 : 8);
#pragma unroll
  for (int nt = 0; nt < 8; ++nt) {
    int nc = nt * 16 + (lane & 15);
    float bv = bias[nc];
#pragma unroll
    for (int i = 0; i < 8; ++i)
      out[(size_t)(mb + i) * kH + nc] = fmaxf(acc[nt][i] + bv, 0.f);
  }
}

// out16 = [relu]( A16[M,128] @ Bp (+bias) )
__global__ void k_gemm_h128(const _Float16* A, const _Float16* Bp, _Float16* out, int M,
                            const float* bias, int relu) {
  int wid = blockIdx.x * (blockDim.x >> 5) + (threadIdx.x >> 5);
  int lane = threadIdx.x & 31;
  if (wid * 16 >= M) return;
  const _Float16* row = A + (size_t)(wid * 16 + (lane & 15)) * kH;
  __builtin_prefetch(row, 0, 3);
  v8f acc[8] = {};
#pragma unroll
  for (int kb = 0; kb < 4; ++kb) {
    v16h a = frag_a_f16(row, kb * 32, lane);
#pragma unroll
    for (int nt = 0; nt < 8; ++nt)
      acc[nt] = wmma32(a, frag_b_packed(Bp, 8, kb, nt, lane), acc[nt]);
  }
  int mb = wid * 16 + ((lane < 16) ? 0 : 8);
#pragma unroll
  for (int nt = 0; nt < 8; ++nt) {
    int nc = nt * 16 + (lane & 15);
    float bv = bias ? bias[nc] : 0.f;
#pragma unroll
    for (int i = 0; i < 8; ++i) {
      float v = acc[nt][i] + bv;
      if (relu) v = fmaxf(v, 0.f);
      out[(size_t)(mb + i) * kH + nc] = (_Float16)v;
    }
  }
}

// out16 = EA[M,32] @ Bp (+bias)   (K=32, single K-step)
__global__ void k_edge_gemm(const float* EA, const _Float16* Bp, const float* bias,
                            _Float16* out, int M) {
  int wid = blockIdx.x * (blockDim.x >> 5) + (threadIdx.x >> 5);
  int lane = threadIdx.x & 31;
  if (wid * 16 >= M) return;
  const float* row = EA + (size_t)(wid * 16 + (lane & 15)) * kEdgeIn;
  v16h a = frag_a_f32(row, 0, lane);
  v8f acc[8] = {};
#pragma unroll
  for (int nt = 0; nt < 8; ++nt)
    acc[nt] = wmma32(a, frag_b_packed(Bp, 8, 0, nt, lane), acc[nt]);
  int mb = wid * 16 + ((lane < 16) ? 0 : 8);
#pragma unroll
  for (int nt = 0; nt < 8; ++nt) {
    int nc = nt * 16 + (lane & 15);
    float bv = bias ? bias[nc] : 0.f;
#pragma unroll
    for (int i = 0; i < 8; ++i)
      out[(size_t)(mb + i) * kH + nc] = (_Float16)(acc[nt][i] + bv);
  }
}

// h += AGG[M,128] @ Wout + b   (attention output projection + residual)
__global__ void k_out_proj(const float* AGG, const _Float16* Bp, const float* bias,
                           float* h, int M) {
  int wid = blockIdx.x * (blockDim.x >> 5) + (threadIdx.x >> 5);
  int lane = threadIdx.x & 31;
  if (wid * 16 >= M) return;
  const float* row = AGG + (size_t)(wid * 16 + (lane & 15)) * kH;
  v8f acc[8] = {};
#pragma unroll
  for (int kb = 0; kb < 4; ++kb) {
    v16h a = frag_a_f32(row, kb * 32, lane);
#pragma unroll
    for (int nt = 0; nt < 8; ++nt)
      acc[nt] = wmma32(a, frag_b_packed(Bp, 8, kb, nt, lane), acc[nt]);
  }
  int mb = wid * 16 + ((lane < 16) ? 0 : 8);
#pragma unroll
  for (int nt = 0; nt < 8; ++nt) {
    int nc = nt * 16 + (lane & 15);
    float bv = bias[nc];
#pragma unroll
    for (int i = 0; i < 8; ++i)
      h[(size_t)(mb + i) * kH + nc] += acc[nt][i] + bv;
  }
}

// Inception temporal conv as shifted GEMMs, + bias + residual(hres) + relu -> f16
// (per-tile mapping kept: column groups have different tap counts)
__global__ void k_conv(const _Float16* nf, const float* hres, const _Float16* convp,
                       const float* b0, const float* b1, const float* b2, const float* b3,
                       _Float16* out16) {
  int wid = blockIdx.x * (blockDim.x >> 5) + (threadIdx.x >> 5);
  int lane = threadIdx.x & 31;
  int mt = wid >> 3, nt = wid & 7;
  if (mt >= kRows / 16) return;
  int r0 = mt * 16;
  int t = r0 / kN;
  int n0 = r0 - t * kN;
  int g = nt >> 1, nthalf = nt & 1;
  const int ksz[4] = {1, 2, 3, 5};
  const int gb[4]  = {0, 1, 3, 6};
  int K = ksz[g];
  v8f c = {};
  int ml = lane & 15;
  for (int j = 0; j < K; ++j) {
    int ts = t - (K - 1 - j) * 2;
    if (ts < 0) continue; // wave-uniform
    const _Float16* row = nf + ((size_t)ts * kN + n0 + ml) * kH;
    const _Float16* Bp = convp + (size_t)(gb[g] + j) * 4096;
#pragma unroll
    for (int kb = 0; kb < 4; ++kb)
      c = wmma32(frag_a_f16(row, kb * 32, lane), frag_b_packed(Bp, 2, kb, nthalf, lane), c);
  }
  const float* bias = (g == 0) ? b0 : (g == 1) ? b1 : (g == 2) ? b2 : b3;
  int nc = nt * 16 + (lane & 15);
  int cb = nc & 31;
  int mb = r0 + ((lane < 16) ? 0 : 8);
  float bv = bias[cb];
#pragma unroll
  for (int i = 0; i < 8; ++i) {
    float v = c[i] + bv + hres[(size_t)(mb + i) * kH + nc];
    out16[(size_t)(mb + i) * kH + nc] = (_Float16)fmaxf(v, 0.f);
  }
}

// cls1: er = [no[src] | no[dst] | ep] (K=384) ; out = relu(er @ Wc1 + b1) -> f16
__global__ void k_cls1(const _Float16* no, int t, const _Float16* ep,
                       const int* src, const int* dst, const _Float16* Bp,
                       const float* bias, _Float16* out) {
  int wid = blockIdx.x * (blockDim.x >> 5) + (threadIdx.x >> 5);
  int lane = threadIdx.x & 31;
  if (wid * 16 >= kE) return;
  int m = wid * 16 + (lane & 15);
  int sm = src[m], dm = dst[m];
  const _Float16* rowS = no + ((size_t)t * kN + sm) * kH;
  const _Float16* rowD = no + ((size_t)t * kN + dm) * kH;
  const _Float16* rowE = ep + (size_t)m * kH;
  v8f acc[8] = {};
#pragma unroll
  for (int ks = 0; ks < 12; ++ks) {
    int seg = ks >> 2;
    int kin = (ks & 3) * 32;
    const _Float16* row = (seg == 0) ? rowS : (seg == 1) ? rowD : rowE;
    v16h a = frag_a_f16(row, kin, lane);
#pragma unroll
    for (int nt = 0; nt < 8; ++nt)
      acc[nt] = wmma32(a, frag_b_packed(Bp, 8, ks, nt, lane), acc[nt]);
  }
  int mb = wid * 16 + ((lane < 16) ? 0 : 8);
#pragma unroll
  for (int nt = 0; nt < 8; ++nt) {
    int nc = nt * 16 + (lane & 15);
    float bv = bias[nc];
#pragma unroll
    for (int i = 0; i < 8; ++i)
      out[(size_t)(mb + i) * kH + nc] = (_Float16)fmaxf(acc[nt][i] + bv, 0.f);
  }
}

// cls2: out[E,5] = h1 @ Wc2 + b2  (N padded to 16, only cols<5 stored)
__global__ void k_cls2(const _Float16* h1, const _Float16* Bp, const float* bias, float* out) {
  int wid = blockIdx.x * (blockDim.x >> 5) + (threadIdx.x >> 5);
  int lane = threadIdx.x & 31;
  if (wid * 16 >= kE) return;
  const _Float16* row = h1 + (size_t)(wid * 16 + (lane & 15)) * kH;
  v8f c = {};
#pragma unroll
  for (int kb = 0; kb < 4; ++kb)
    c = wmma32(frag_a_f16(row, kb * 32, lane), frag_b_packed(Bp, 1, kb, 0, lane), c);
  int nc = lane & 15;
  int mb = wid * 16 + ((lane < 16) ? 0 : 8);
  if (nc < kNC) {
    float bv = bias[nc];
#pragma unroll
    for (int i = 0; i < 8; ++i)
      out[(size_t)(mb + i) * kNC + nc] = c[i] + bv;
  }
}

// ---------------- normalization kernels ----------------
__global__ void k_layernorm(const float* h, const float* g, const float* b, _Float16* out) {
  int row = blockIdx.x, c = threadIdx.x; // 128 threads
  __shared__ float sh[128];
  float x = h[(size_t)row * kH + c];
  sh[c] = x; __syncthreads();
  for (int s = 64; s > 0; s >>= 1) { if (c < s) sh[c] += sh[c + s]; __syncthreads(); }
  float mu = sh[0] * (1.f / kH);
  __syncthreads();
  float d = x - mu;
  sh[c] = d * d; __syncthreads();
  for (int s = 64; s > 0; s >>= 1) { if (c < s) sh[c] += sh[c + s]; __syncthreads(); }
  float var = sh[0] * (1.f / kH);
  out[(size_t)row * kH + c] = (_Float16)(d * rsqrtf(var + 1e-5f) * g[c] + b[c]);
}

__global__ void k_bn_stats(const float* h, float* stats) {
  int c = blockIdx.x, t = blockIdx.y, tid = threadIdx.x; // 256 threads
  float s = 0.f, s2 = 0.f;
  for (int n = tid; n < kN; n += 256) {
    float v = h[((size_t)t * kN + n) * kH + c];
    s += v; s2 += v * v;
  }
  __shared__ float sh[256], sh2[256];
  sh[tid] = s; sh2[tid] = s2; __syncthreads();
  for (int st = 128; st > 0; st >>= 1) {
    if (tid < st) { sh[tid] += sh[tid + st]; sh2[tid] += sh2[tid + st]; }
    __syncthreads();
  }
  if (tid == 0) {
    float mu = sh[0] / kN;
    stats[(t * kH + c) * 2]     = mu;
    stats[(t * kH + c) * 2 + 1] = sh2[0] / kN - mu * mu;
  }
}

__global__ void k_bn_apply(float* h, const float* stats, const float* g, const float* b,
                           _Float16* out16, long long total) {
  long long i = (long long)blockIdx.x * blockDim.x + threadIdx.x;
  if (i >= total) return;
  int c = (int)(i & 127);
  int t = (int)((i >> 7) / kN);
  float mu = stats[(t * kH + c) * 2];
  float var = stats[(t * kH + c) * 2 + 1];
  float v = fmaxf((h[i] - mu) * rsqrtf(var + 1e-5f) * g[c] + b[c], 0.f);
  h[i] = v;
  if (out16) out16[i] = (_Float16)v;
}

// ---------------- attention edge kernels ----------------
__device__ __forceinline__ unsigned f2ord(float f) {
  unsigned u = __float_as_uint(f);
  return (u & 0x80000000u) ? ~u : (u | 0x80000000u);
}
__device__ __forceinline__ float ord2f(unsigned u) {
  return (u & 0x80000000u) ? __uint_as_float(u & 0x7FFFFFFFu) : __uint_as_float(~u);
}

__global__ void k_logits(const _Float16* q, const _Float16* kk, const _Float16* e,
                         const int* src, const int* dst, float* ab, unsigned* mbuf) {
  int tid = blockIdx.x * blockDim.x + threadIdx.x;
  if (tid >= kE * kNH) return;
  int eid = tid >> 2, hh = tid & 3;
  int s = src[eid], d = dst[eid];
  const _Float16* qp = q + (size_t)d * kH + hh * 32;
  const _Float16* kp = kk + (size_t)s * kH + hh * 32;
  const _Float16* epp = e + (size_t)eid * kH + hh * 32;
  float acc = 0.f;
#pragma unroll
  for (int i = 0; i < 32; ++i)
    acc += (float)qp[i] * ((float)kp[i] + (float)epp[i]);
  acc *= kInvSqrtHD;
  ab[tid] = acc;
  atomicMax(&mbuf[(size_t)d * kNH + hh], f2ord(acc));
}

__global__ void k_expsum(float* ab, const int* dst, const unsigned* mbuf, float* sbuf) {
  int tid = blockIdx.x * blockDim.x + threadIdx.x;
  if (tid >= kE * kNH) return;
  int eid = tid >> 2, hh = tid & 3;
  int d = dst[eid];
  float a = __expf(ab[tid] - ord2f(mbuf[(size_t)d * kNH + hh]));
  ab[tid] = a;
  atomicAdd(&sbuf[(size_t)d * kNH + hh], a);
}

__global__ void k_agg(const float* ab, const _Float16* vv, const _Float16* e,
                      const int* src, const int* dst, const float* sbuf, float* agg) {
  long long tid = (long long)blockIdx.x * blockDim.x + threadIdx.x;
  if (tid >= (long long)kE * kH) return;
  int eid = (int)(tid >> 7), c = (int)(tid & 127);
  int hh = c >> 5;
  int d = dst[eid], s = src[eid];
  float a = ab[(size_t)eid * kNH + hh] / (sbuf[(size_t)d * kNH + hh] + 1e-16f);
  float val = ((float)vv[(size_t)s * kH + c] + (float)e[(size_t)eid * kH + c]) * a;
  atomicAdd(&agg[(size_t)d * kH + c], val);
}

// ---------------- input mapping ----------------
// params pytree (sorted keys), 36 leaves:
// bn1{b,g} bn2{b,g} cls1{b,w} cls2{b,w} edge_proj{b,w}
// gnn1{ln_b,ln_g,out{b,w},we,wk,wq,wv} gnn2{...} incep[0..3]{b,w} node_enc{b,w}
enum {
  P_BN1_B = 0, P_BN1_G, P_BN2_B, P_BN2_G,
  P_CLS1_B, P_CLS1_W, P_CLS2_B, P_CLS2_W, P_EP_B, P_EP_W,
  P_G1_LNB, P_G1_LNG, P_G1_OUTB, P_G1_OUTW, P_G1_WE, P_G1_WK, P_G1_WQ, P_G1_WV,
  P_G2_LNB, P_G2_LNG, P_G2_OUTB, P_G2_OUTW, P_G2_WE, P_G2_WK, P_G2_WQ, P_G2_WV,
  P_I0_B, P_I0_W, P_I1_B, P_I1_W, P_I2_B, P_I2_W, P_I3_B, P_I3_W,
  P_NE_B, P_NE_W,
  P_COUNT // == 36
};

struct Inputs {
  const float* x;
  const float* ea;
  const int* ei;
  const float* prm[P_COUNT];
};

extern "C" void kernel_launch(void* const* d_in, const int* in_sizes, int n_in,
                              void* d_out, int out_size, void* d_ws, size_t ws_size,
                              hipStream_t stream) {
  (void)n_in; (void)out_size;
  Inputs I;
  if (in_sizes[0] == kT * kN * kNodeIn) {        // insertion-order top level
    I.x = (const float*)d_in[0];
    I.ea = (const float*)d_in[1];
    for (int i = 0; i < P_COUNT; ++i) I.prm[i] = (const float*)d_in[2 + i];
    I.ei = (const int*)d_in[2 + P_COUNT];        // 38
  } else {                                        // fully sorted pytree order
    I.ea = (const float*)d_in[0];
    I.ei = (const int*)d_in[1];
    for (int i = 0; i < P_COUNT; ++i) I.prm[i] = (const float*)d_in[3 + i];
    I.x = (const float*)d_in[3 + P_COUNT];       // 39
  }
  float* outp = (float*)d_out;

  // -------- workspace layout --------
  char* ws = (char*)d_ws;
  size_t off = 0;
  auto alloc = [&](size_t bytes) -> char* {
    char* p = ws + off;
    off = (off + bytes + 255) & ~(size_t)255;
    return p;
  };
  _Float16* wenc = (_Float16*)alloc(8192 * 2);
  _Float16* wq[2]; _Float16* wk[2]; _Float16* wv[2]; _Float16* wout[2]; _Float16* we[2];
  for (int l = 0; l < 2; ++l) {
    wq[l]   = (_Float16*)alloc(16384 * 2);
    wk[l]   = (_Float16*)alloc(16384 * 2);
    wv[l]   = (_Float16*)alloc(16384 * 2);
    wout[l] = (_Float16*)alloc(16384 * 2);
    we[l]   = (_Float16*)alloc(4096 * 2);
  }
  _Float16* wep   = (_Float16*)alloc(4096 * 2);
  _Float16* wc1   = (_Float16*)alloc(49152 * 2);
  _Float16* wc2   = (_Float16*)alloc(2048 * 2);
  _Float16* wconv = (_Float16*)alloc(45056 * 2);
  float*    hbuf  = (float*)alloc((size_t)kRows * kH * 4);
  _Float16* f16a  = (_Float16*)alloc((size_t)kRows * kH * 2);
  _Float16* f16b  = (_Float16*)alloc((size_t)kRows * kH * 2);
  _Float16* q16   = (_Float16*)alloc((size_t)kRows * kH * 2);
  _Float16* k16   = (_Float16*)alloc((size_t)kRows * kH * 2);
  _Float16* v16   = (_Float16*)alloc((size_t)kRows * kH * 2);
  float*    agg   = (float*)alloc((size_t)kRows * kH * 4);
  _Float16* e16   = (_Float16*)alloc((size_t)kE * kH * 2);
  _Float16* h1    = (_Float16*)alloc((size_t)kE * kH * 2);
  float*    abuf  = (float*)alloc((size_t)kE * kNH * 4);
  unsigned* mbuf  = (unsigned*)alloc((size_t)kN * kNH * 4);
  float*    sbuf  = (float*)alloc((size_t)kN * kNH * 4);
  float*    stats = (float*)alloc((size_t)kT * kH * 2 * 4);
  if (off > ws_size) return;

  const int ksz[4] = {1, 2, 3, 5};
  const int gb[4]  = {0, 1, 3, 6};

  // -------- pack weights --------
  k_pack_b<<<32, 256, 0, stream>>>(I.prm[P_NE_W], wenc, 64, 128, 2, 8);
  const int WQ[2] = {P_G1_WQ, P_G2_WQ}, WKi[2] = {P_G1_WK, P_G2_WK},
            WVi[2] = {P_G1_WV, P_G2_WV}, WO[2] = {P_G1_OUTW, P_G2_OUTW},
            WEi[2] = {P_G1_WE, P_G2_WE};
  for (int l = 0; l < 2; ++l) {
    k_pack_b<<<64, 256, 0, stream>>>(I.prm[WQ[l]],  wq[l],   128, 128, 4, 8);
    k_pack_b<<<64, 256, 0, stream>>>(I.prm[WKi[l]], wk[l],   128, 128, 4, 8);
    k_pack_b<<<64, 256, 0, stream>>>(I.prm[WVi[l]], wv[l],   128, 128, 4, 8);
    k_pack_b<<<64, 256, 0, stream>>>(I.prm[WO[l]],  wout[l], 128, 128, 4, 8);
    k_pack_b<<<16, 256, 0, stream>>>(I.prm[WEi[l]], we[l],   32,  128, 1, 8);
  }
  k_pack_b<<<16, 256, 0, stream>>>(I.prm[P_EP_W], wep, 32, 128, 1, 8);
  k_pack_b<<<192, 256, 0, stream>>>(I.prm[P_CLS1_W], wc1, 384, 128, 12, 8);
  k_pack_b<<<8, 256, 0, stream>>>(I.prm[P_CLS2_W], wc2, 128, 5, 4, 1);
  const int IW[4] = {P_I0_W, P_I1_W, P_I2_W, P_I3_W};
  for (int g = 0; g < 4; ++g)
    for (int j = 0; j < ksz[g]; ++j)
      k_pack_conv<<<16, 256, 0, stream>>>(I.prm[IW[g]], wconv + (size_t)(gb[g] + j) * 4096,
                                          ksz[g], j);

  // -------- node encoder --------
  k_node_enc<<<625, 256, 0, stream>>>(I.x, wenc, I.prm[P_NE_B], hbuf, kRows);

  // -------- two graph-transformer layers --------
  const int LNG[2] = {P_G1_LNG, P_G2_LNG}, LNB[2] = {P_G1_LNB, P_G2_LNB},
            OB[2]  = {P_G1_OUTB, P_G2_OUTB};
  const int bnG[2] = {P_BN1_G, P_BN2_G};
  const int bnB[2] = {P_BN1_B, P_BN2_B};

  for (int l = 0; l < 2; ++l) {
    k_layernorm<<<kRows, 128, 0, stream>>>(hbuf, I.prm[LNG[l]], I.prm[LNB[l]], f16a);
    k_gemm_h128<<<625, 256, 0, stream>>>(f16a, wq[l], q16, kRows, nullptr, 0);
    k_gemm_h128<<<625, 256, 0, stream>>>(f16a, wk[l], k16, kRows, nullptr, 0);
    k_gemm_h128<<<625, 256, 0, stream>>>(f16a, wv[l], v16, kRows, nullptr, 0);
    k_fill<<<40000, 256, 0, stream>>>((unsigned*)agg, 0u, (long long)kRows * kH);
    for (int t = 0; t < kT; ++t) {
      const int* src = I.ei + (size_t)t * 2 * kE;
      const int* dst = src + kE;
      k_fill<<<157, 256, 0, stream>>>(mbuf, 0u, (long long)kN * kNH);
      k_fill<<<157, 256, 0, stream>>>((unsigned*)sbuf, 0u, (long long)kN * kNH);
      k_edge_gemm<<<469, 256, 0, stream>>>(I.ea + (size_t)t * kE * kEdgeIn, we[l],
                                           nullptr, e16, kE);
      k_logits<<<938, 256, 0, stream>>>(q16 + (size_t)t * kN * kH, k16 + (size_t)t * kN * kH,
                                        e16, src, dst, abuf, mbuf);
      k_expsum<<<938, 256, 0, stream>>>(abuf, dst, mbuf, sbuf);
      k_agg<<<30000, 256, 0, stream>>>(abuf, v16 + (size_t)t * kN * kH, e16, src, dst, sbuf,
                                       agg + (size_t)t * kN * kH);
    }
    k_out_proj<<<625, 256, 0, stream>>>(agg, wout[l], I.prm[OB[l]], hbuf, kRows);
    k_bn_stats<<<dim3(kH, kT), 256, 0, stream>>>(hbuf, stats);
    k_bn_apply<<<40000, 256, 0, stream>>>(hbuf, stats, I.prm[bnG[l]], I.prm[bnB[l]],
                                          (l == 1) ? f16a : (_Float16*)nullptr,
                                          (long long)kRows * kH);
  }

  // -------- temporal inception conv + residual + relu --------
  k_conv<<<5000, 256, 0, stream>>>(f16a, hbuf, wconv, I.prm[P_I0_B], I.prm[P_I1_B],
                                   I.prm[P_I2_B], I.prm[P_I3_B], f16b);

  // -------- per-frame edge classifier --------
  for (int t = 0; t < kT; ++t) {
    const int* src = I.ei + (size_t)t * 2 * kE;
    const int* dst = src + kE;
    k_edge_gemm<<<469, 256, 0, stream>>>(I.ea + (size_t)t * kE * kEdgeIn, wep,
                                         I.prm[P_EP_B], e16, kE);
    k_cls1<<<469, 256, 0, stream>>>(f16b, t, e16, src, dst, wc1, I.prm[P_CLS1_B], h1);
    k_cls2<<<469, 256, 0, stream>>>(h1, wc2, I.prm[P_CLS2_B],
                                    outp + (size_t)t * kE * kNC);
  }
}
